// AutoEncoderTopKnotriton_24910810317237
// MI455X (gfx1250) — matrix operand order repaired and checked by reading, not compiled, not run on hardware
//
#include <hip/hip_runtime.h>
#include <hip/hip_bf16.h>
#include <cstdint>

// ---------------------------------------------------------------------------
// AutoEncoderTopK on MI455X (gfx1250)
//   B=2048, A=4096, D=32768, K=4096
//   encode GEMM: bf16 WMMA (v_wmma_f32_16x16x32_bf16), double-buffered
//                software-pipelined async global->LDS staging
//   global top-K: 3-level radix-select on float bits + deterministic tie-break
//   decode: per-row register accumulation (no atomics)
// ---------------------------------------------------------------------------

#define B_  2048
#define A_  4096
#define D_  32768
#define K_  4096
#define ND  ((size_t)B_ * (size_t)D_)     // 67108864 latents

typedef __bf16 v16bf __attribute__((ext_vector_type(16)));
typedef __bf16 bf16x4 __attribute__((ext_vector_type(4)));
typedef float  v8f   __attribute__((ext_vector_type(8)));

static __device__ __forceinline__ v8f wmma_bf16(v16bf a, v16bf b, v8f c) {
  // (neg_a, A, neg_b, B, c_mod, C, reuse_a, reuse_b)
  return __builtin_amdgcn_wmma_f32_16x16x32_bf16(false, a, false, b, (short)0, c,
                                                 false, false);
}

// ---------------------------------------------------------------------------
// 0) zero the small selection buffers (ws is poisoned 0xAA once by harness)
// ---------------------------------------------------------------------------
__global__ void k_zero(unsigned int* p, int n) {
  int i = blockIdx.x * blockDim.x + threadIdx.x;
  if (i < n) p[i] = 0u;
}

// ---------------------------------------------------------------------------
// 1) W_enc (fp32) -> bf16
// ---------------------------------------------------------------------------
__global__ void k_conv_w(const float* __restrict__ W, __bf16* __restrict__ Wb) {
  size_t i4 = (size_t)blockIdx.x * blockDim.x + threadIdx.x;   // group of 4
  size_t e = i4 * 4;
  float4 v = *(const float4*)(W + e);
  bf16x4 o;
  o[0] = (__bf16)v.x; o[1] = (__bf16)v.y; o[2] = (__bf16)v.z; o[3] = (__bf16)v.w;
  *(bf16x4*)(Wb + e) = o;
}

// ---------------------------------------------------------------------------
// 2) xb = bf16(x - b_dec)  (fuses the encoder's input bias subtraction)
// ---------------------------------------------------------------------------
__global__ void k_conv_x(const float* __restrict__ X, const float* __restrict__ bdec,
                         __bf16* __restrict__ Xb) {
  size_t i4 = (size_t)blockIdx.x * blockDim.x + threadIdx.x;
  size_t e = i4 * 4;
  int col = (int)(e & (A_ - 1));
  float4 v = *(const float4*)(X + e);
  float4 b = *(const float4*)(bdec + col);
  bf16x4 o;
  o[0] = (__bf16)(v.x - b.x); o[1] = (__bf16)(v.y - b.y);
  o[2] = (__bf16)(v.z - b.z); o[3] = (__bf16)(v.w - b.w);
  *(bf16x4*)(Xb + e) = o;
}

// ---------------------------------------------------------------------------
// 3) GEMM: Z[B,D] = relu(Xb @ Wb^T + b_enc)
//    block tile 128x128, K-chunk 32, 8 waves (2 M x 4 N), wave tile 64x32.
//    Double-buffered LDS; async copies for tile t+1 overlap WMMA on tile t.
//    Pitch 40 bf16 = 80B rows: 16B-aligned b128 loads, conflict-free banks
//    (20-DWORD stride, gcd(20,64)=4 -> distinct banks over 16 rows).
// ---------------------------------------------------------------------------
#define BKC 32
#define LDS_PITCH 40
#define TILE_BYTES (128 * LDS_PITCH * 2)

__global__ __launch_bounds__(256) void k_gemm(const __bf16* __restrict__ Xb,
                                              const __bf16* __restrict__ Wb,
                                              const float* __restrict__ benc,
                                              float* __restrict__ Z) {
  __shared__ __bf16 xs[2][128][LDS_PITCH];
  __shared__ __bf16 wsm[2][128][LDS_PITCH];

  const int tid  = threadIdx.x;
  const int m0   = blockIdx.x * 128;   // grid.x fastest -> N-stripe L2 reuse of W
  const int n0   = blockIdx.y * 128;
  const int wave = tid >> 5;
  const int lane = tid & 31;
  const int r    = lane & 15;          // row-within-fragment / column id
  const int h    = lane >> 4;          // half-wave
  const int wm   = wave >> 2;          // 0..1 (M)
  const int wn   = wave & 3;           // 0..3 (N)

  v8f acc[4][2];
  const v8f vzero = {0.f, 0.f, 0.f, 0.f, 0.f, 0.f, 0.f, 0.f};
#pragma unroll
  for (int m = 0; m < 4; ++m) { acc[m][0] = vzero; acc[m][1] = vzero; }

  const uint32_t xsBase = (uint32_t)(uintptr_t)&xs[0][0][0];
  const uint32_t wsBase = (uint32_t)(uintptr_t)&wsm[0][0][0];

  // issue one 128x32 bf16 tile pair (x + w) into LDS buffer `buf`
  // 512 16B chunks per tile -> 2 per thread per tile -> 4 async ops per thread
  auto issue_tiles = [&](int kt, int buf) {
    const int k0 = kt * BKC;
#pragma unroll
    for (int i = 0; i < 2; ++i) {
      const int chunk = tid + 256 * i;        // 0..511
      const int row   = chunk >> 2;           // 0..127
      const int c     = chunk & 3;            // 8-bf16 sub-chunk
      const uint32_t ldsx = xsBase + buf * TILE_BYTES + row * (LDS_PITCH * 2) + c * 16;
      const uint64_t gx =
          (uint64_t)(uintptr_t)(Xb + (size_t)(m0 + row) * A_ + k0 + c * 8);
      asm volatile("global_load_async_to_lds_b128 %0, %1, off"
                   :: "v"(ldsx), "v"(gx) : "memory");
      const uint32_t ldsw = wsBase + buf * TILE_BYTES + row * (LDS_PITCH * 2) + c * 16;
      const uint64_t gw =
          (uint64_t)(uintptr_t)(Wb + (size_t)(n0 + row) * A_ + k0 + c * 8);
      asm volatile("global_load_async_to_lds_b128 %0, %1, off"
                   :: "v"(ldsw), "v"(gw) : "memory");
    }
  };

  issue_tiles(0, 0);

  const int NT = A_ / BKC;   // 128
  for (int kt = 0; kt < NT; ++kt) {
    const int cur = kt & 1;
    if (kt + 1 < NT) {
      issue_tiles(kt + 1, cur ^ 1);
      // drain only the OLDER 4 async ops (this wave's copies into `cur`);
      // per-wave async ops complete in order, so <=4 outstanding leaves the
      // prefetch in flight while we compute.
      asm volatile("s_wait_asynccnt 0x4" ::: "memory");
    } else {
      asm volatile("s_wait_asynccnt 0x0" ::: "memory");
    }
    __syncthreads();

    // B fragments: lane holds contiguous 16 K of column r (same K order as A)
    v16bf bf[2];
#pragma unroll
    for (int n = 0; n < 2; ++n) {
      const int br = wn * 32 + n * 16 + r;
      const uint4* p = (const uint4*)&wsm[cur][br][h * 16];
      ((uint4*)&bf[n])[0] = p[0];
      ((uint4*)&bf[n])[1] = p[1];
    }
#pragma unroll
    for (int m = 0; m < 4; ++m) {
      const int ar = wm * 64 + m * 16 + r;
      v16bf af;   // A: contiguous K {h*16..h*16+15}
      ((uint4*)&af)[0] = *(const uint4*)&xs[cur][ar][h * 16];
      ((uint4*)&af)[1] = *(const uint4*)&xs[cur][ar][h * 16 + 8];
      acc[m][0] = wmma_bf16(af, bf[0], acc[m][0]);
      acc[m][1] = wmma_bf16(af, bf[1], acc[m][1]);
    }

    // protects buffer `cur` from being overwritten by next iteration's prefetch
    __syncthreads();
  }

  // epilogue: C/D layout -> M = j + h*8, N = r
#pragma unroll
  for (int n = 0; n < 2; ++n) {
    const int col = n0 + wn * 32 + n * 16 + r;
    const float be = benc[col];
#pragma unroll
    for (int m = 0; m < 4; ++m) {
#pragma unroll
      for (int j = 0; j < 8; ++j) {
        const int row = m0 + wm * 64 + m * 16 + h * 8 + j;
        float v = acc[m][n][j] + be;
        v = v > 0.f ? v : 0.f;
        Z[(size_t)row * D_ + col] = v;
      }
    }
  }
}

// ---------------------------------------------------------------------------
// 4-8) exact top-K threshold via 3-level radix select on float bits
//      (z >= 0 after relu -> float bits are monotonic)
// ---------------------------------------------------------------------------
__global__ void k_hist1(const float* __restrict__ Z, unsigned int* __restrict__ hist) {
  __shared__ unsigned int lh[4096];
  for (int i = threadIdx.x; i < 4096; i += blockDim.x) lh[i] = 0u;
  __syncthreads();
  for (size_t i = (size_t)blockIdx.x * blockDim.x + threadIdx.x; i < ND;
       i += (size_t)gridDim.x * blockDim.x) {
    unsigned int key = __float_as_uint(Z[i]);
    atomicAdd(&lh[key >> 20], 1u);
  }
  __syncthreads();
  for (int i = threadIdx.x; i < 4096; i += blockDim.x)
    if (lh[i]) atomicAdd(&hist[i], lh[i]);
}

__global__ void k_hist2(const float* __restrict__ Z, const unsigned int* __restrict__ scal,
                        unsigned int* __restrict__ hist) {
  __shared__ unsigned int lh[4096];
  for (int i = threadIdx.x; i < 4096; i += blockDim.x) lh[i] = 0u;
  __syncthreads();
  const unsigned int b1 = scal[0];
  for (size_t i = (size_t)blockIdx.x * blockDim.x + threadIdx.x; i < ND;
       i += (size_t)gridDim.x * blockDim.x) {
    unsigned int key = __float_as_uint(Z[i]);
    if ((key >> 20) == b1) atomicAdd(&lh[(key >> 8) & 4095u], 1u);
  }
  __syncthreads();
  for (int i = threadIdx.x; i < 4096; i += blockDim.x)
    if (lh[i]) atomicAdd(&hist[i], lh[i]);
}

__global__ void k_hist3(const float* __restrict__ Z, const unsigned int* __restrict__ scal,
                        unsigned int* __restrict__ hist) {
  __shared__ unsigned int lh[256];
  for (int i = threadIdx.x; i < 256; i += blockDim.x) lh[i] = 0u;
  __syncthreads();
  const unsigned int top24 = (scal[0] << 12) | scal[2];
  for (size_t i = (size_t)blockIdx.x * blockDim.x + threadIdx.x; i < ND;
       i += (size_t)gridDim.x * blockDim.x) {
    unsigned int key = __float_as_uint(Z[i]);
    if ((key >> 8) == top24) atomicAdd(&lh[key & 255u], 1u);
  }
  __syncthreads();
  for (int i = threadIdx.x; i < 256; i += blockDim.x)
    if (lh[i]) atomicAdd(&hist[i], lh[i]);
}

// pick bin b with (count strictly above b) < K <= (count >= b)
__global__ void k_select(const unsigned int* __restrict__ hist,
                         unsigned int* __restrict__ scal, int level, int nbins) {
  unsigned int base = (level == 1) ? 0u : ((level == 2) ? scal[1] : scal[3]);
  unsigned int acc = base;
  int bsel = 0;
  for (int b = nbins - 1; b >= 0; --b) {
    unsigned int c = hist[b];
    if (acc + c >= (unsigned int)K_) { bsel = b; break; }
    acc += c;
  }
  if (level == 1)      { scal[0] = (unsigned int)bsel; scal[1] = acc; }
  else if (level == 2) { scal[2] = (unsigned int)bsel; scal[3] = acc; }
  else {
    scal[4] = (scal[0] << 20) | (scal[2] << 8) | (unsigned int)bsel; // exact v bits
    scal[5] = acc;                         // count strictly greater than v
    scal[6] = (unsigned int)K_ - acc;      // how many == v to take (lowest index)
  }
}

// ---------------------------------------------------------------------------
// 9) compaction: indices with key > v (fixed set), and key == v candidates
// ---------------------------------------------------------------------------
#define EQ_CAP 8192
__global__ void k_compact(const float* __restrict__ Z, const unsigned int* __restrict__ scal,
                          unsigned int* __restrict__ cnt, unsigned int* __restrict__ gt,
                          unsigned int* __restrict__ eq) {
  const unsigned int v = scal[4];
  for (size_t i = (size_t)blockIdx.x * blockDim.x + threadIdx.x; i < ND;
       i += (size_t)gridDim.x * blockDim.x) {
    unsigned int key = __float_as_uint(Z[i]);
    if (key > v) {
      unsigned int p = atomicAdd(&cnt[0], 1u);
      if (p < (unsigned int)K_) gt[p] = (unsigned int)i;
    } else if (key == v) {
      unsigned int p = atomicAdd(&cnt[1], 1u);
      if (p < (unsigned int)EQ_CAP) eq[p] = (unsigned int)i;
    }
  }
}

// ---------------------------------------------------------------------------
// 10) finalize: sort ==v candidates, take lowest-index `need`, sort final K.
//     Single block; makes the whole selection deterministic.
// ---------------------------------------------------------------------------
static __device__ void bitonic_sort(unsigned int* sb, unsigned int n) {
  for (unsigned int k = 2; k <= n; k <<= 1) {
    for (unsigned int j = k >> 1; j >= 1; j >>= 1) {
      for (unsigned int i = threadIdx.x; i < n; i += blockDim.x) {
        unsigned int l = i ^ j;
        if (l > i) {
          bool up = ((i & k) == 0);
          unsigned int a = sb[i], b = sb[l];
          if ((a > b) == up) { sb[i] = b; sb[l] = a; }
        }
      }
      __syncthreads();
    }
  }
}

__global__ __launch_bounds__(1024) void k_finalize(const unsigned int* __restrict__ scal,
                                                   const unsigned int* __restrict__ cnt,
                                                   unsigned int* __restrict__ gt,
                                                   const unsigned int* __restrict__ eq,
                                                   unsigned int* __restrict__ sel) {
  __shared__ unsigned int sb[EQ_CAP];
  const unsigned int cgt  = scal[5];
  const unsigned int need = scal[6];
  unsigned int neq = cnt[1];
  if (neq > (unsigned int)EQ_CAP) neq = (unsigned int)EQ_CAP;
  unsigned int np = 1;
  while (np < neq) np <<= 1;
  for (unsigned int i = threadIdx.x; i < np; i += blockDim.x)
    sb[i] = (i < neq) ? eq[i] : 0xFFFFFFFFu;
  __syncthreads();
  bitonic_sort(sb, np);
  for (unsigned int i = threadIdx.x; i < need; i += blockDim.x) gt[cgt + i] = sb[i];
  __syncthreads();
  for (unsigned int i = threadIdx.x; i < (unsigned int)K_; i += blockDim.x) sb[i] = gt[i];
  __syncthreads();
  bitonic_sort(sb, (unsigned int)K_);
  for (unsigned int i = threadIdx.x; i < (unsigned int)K_; i += blockDim.x) sel[i] = sb[i];
}

// ---------------------------------------------------------------------------
// 11) decode: one block per batch row; binary-search the sorted selection,
//     accumulate act * W_dec[d] over the row's segment, on top of b_dec.
// ---------------------------------------------------------------------------
static __device__ __forceinline__ unsigned int lower_bound(const unsigned int* a,
                                                           unsigned int n,
                                                           unsigned int v) {
  unsigned int lo = 0, hi = n;
  while (lo < hi) {
    unsigned int m = (lo + hi) >> 1;
    if (a[m] < v) lo = m + 1; else hi = m;
  }
  return lo;
}

__global__ __launch_bounds__(256) void k_decode(const float* __restrict__ Z,
                                                const unsigned int* __restrict__ sel,
                                                const float* __restrict__ Wdec,
                                                const float* __restrict__ bdec,
                                                float* __restrict__ out) {
  const int row = blockIdx.x;
  const int tid = threadIdx.x;
  float acc[16];
#pragma unroll
  for (int i = 0; i < 16; ++i) acc[i] = bdec[i * 256 + tid];

  const unsigned int lob = lower_bound(sel, (unsigned int)K_, (unsigned int)row << 15);
  const unsigned int hib = lower_bound(sel, (unsigned int)K_, (unsigned int)(row + 1) << 15);
  for (unsigned int k = lob; k < hib; ++k) {
    const unsigned int f = sel[k];
    const unsigned int d = f & (D_ - 1);
    const float a = Z[f];                       // act (already relu'd)
    const float* wr = Wdec + (size_t)d * A_;
#pragma unroll
    for (int i = 0; i < 16; ++i) acc[i] += a * wr[i * 256 + tid];
  }
#pragma unroll
  for (int i = 0; i < 16; ++i) out[(size_t)row * A_ + i * 256 + tid] = acc[i];
}

// ---------------------------------------------------------------------------
// launch
// ---------------------------------------------------------------------------
extern "C" void kernel_launch(void* const* d_in, const int* in_sizes, int n_in,
                              void* d_out, int out_size, void* d_ws, size_t ws_size,
                              hipStream_t stream) {
  const float* x     = (const float*)d_in[0];
  const float* W_enc = (const float*)d_in[1];
  const float* b_enc = (const float*)d_in[2];
  const float* W_dec = (const float*)d_in[3];
  const float* b_dec = (const float*)d_in[4];
  float* out = (float*)d_out;

  // workspace layout (requires ~528 MB)
  char* ws = (char*)d_ws;
  __bf16* Wb = (__bf16*)(ws);                                       // 256 MB
  __bf16* Xb = (__bf16*)(ws + 268435456ull);                        //  16 MB
  float*  Z  = (float*)(ws + 285212672ull);                         // 256 MB
  unsigned int* H1   = (unsigned int*)(ws + 553648128ull);          // 4096
  unsigned int* H2   = (unsigned int*)(ws + 553664512ull);          // 4096
  unsigned int* H3   = (unsigned int*)(ws + 553680896ull);          // 256
  unsigned int* SCAL = (unsigned int*)(ws + 553681920ull);          // 16
  unsigned int* CNT  = (unsigned int*)(ws + 553681984ull);          // 16
  unsigned int* GT   = (unsigned int*)(ws + 553682048ull);          // 4096
  unsigned int* EQ   = (unsigned int*)(ws + 553698432ull);          // 8192
  unsigned int* SEL  = (unsigned int*)(ws + 553731200ull);          // 4096

  // 0) zero hist/scal/cnt region (8480 words, contiguous from H1)
  k_zero<<<34, 256, 0, stream>>>(H1, 8480);

  // 1-2) precision conversion (bf16), fusing x - b_dec
  k_conv_w<<<(D_ * (size_t)A_ / 4 + 255) / 256, 256, 0, stream>>>(W_enc, Wb);
  k_conv_x<<<(B_ * (size_t)A_ / 4 + 255) / 256, 256, 0, stream>>>(x, b_dec, Xb);

  // 3) encode GEMM (M tiles fastest for L2 stripe reuse of W)
  dim3 ggrid(B_ / 128, D_ / 128);
  k_gemm<<<ggrid, 256, 0, stream>>>(Xb, Wb, b_enc, Z);

  // 4-8) radix-select exact threshold
  k_hist1<<<8192, 256, 0, stream>>>(Z, H1);
  k_select<<<1, 1, 0, stream>>>(H1, SCAL, 1, 4096);
  k_hist2<<<8192, 256, 0, stream>>>(Z, SCAL, H2);
  k_select<<<1, 1, 0, stream>>>(H2, SCAL, 2, 4096);
  k_hist3<<<8192, 256, 0, stream>>>(Z, SCAL, H3);
  k_select<<<1, 1, 0, stream>>>(H3, SCAL, 3, 256);

  // 9-10) compact + deterministic tie-break + sort of final K indices
  k_compact<<<8192, 256, 0, stream>>>(Z, SCAL, CNT, GT, EQ);
  k_finalize<<<1, 1024, 0, stream>>>(SCAL, CNT, GT, EQ, SEL);

  // 11) sparse decode
  k_decode<<<B_, 256, 0, stream>>>(Z, SEL, W_dec, b_dec, out);
}
